// Block_51221779972150
// MI455X (gfx1250) — matrix operand order repaired and checked by reading, not compiled, not run on hardware
//
#include <hip/hip_runtime.h>
#include <hip/hip_bf16.h>

// ---------------- CDNA5 WMMA plumbing ----------------
typedef __attribute__((ext_vector_type(16))) _Float16 v16h;
typedef __attribute__((ext_vector_type(8)))  float    v8f;

union FragH { v16h v; unsigned u[8]; _Float16 h[16]; };
union FragF { v8f  v; float f[8]; };

__device__ __forceinline__ v8f wmma_f16(v16h a, v16h b, v8f c) {
    return __builtin_amdgcn_wmma_f32_16x16x32_f16(
        false, a, false, b, (short)0, c, false, false);
}

// A-matrix (16x32 f16): lane m=lane&15, half=lane>>4.
//   VGPR v holds K=2v+16h -> dword v+8h
__device__ __forceinline__ int a_dw(int v, int half) { return v + 8 * half; }
// B-matrix (32x16 f16): lane n=lane&15, half=lane>>4, VGPR v holds K=2v+16h -> dword v+8h
__device__ __forceinline__ int b_dw(int v, int half) { return v + 8 * half; }

// ---------------- problem constants ----------------
#define Bc   4
#define Tt   1296
#define Ee   768
#define Ff   3072
#define Rr   (Bc * Tt)   // 5184
#define Gg   36
#define Hh   12
#define HDd  64
#define QKVN 2304        // fused q|k|v width
#define ATTN_SCALE 0.03608439182435161f  // 768^-0.5

// ---------------- conversion / layout kernels ----------------
__global__ void k_patchify(const float* __restrict__ x, _Float16* __restrict__ ape, int n) {
    int idx = blockIdx.x * 256 + threadIdx.x;
    if (idx >= n) return;
    int m = idx / 768, k = idx - m * 768;
    int b = m / Tt, t = m - b * Tt;
    int i = t / Gg, j = t - i * Gg;
    int c = k >> 8, p = (k >> 4) & 15, q = k & 15;
    size_t src = (((size_t)(b * 3 + c) * 576) + (size_t)(i * 16 + p)) * 576 + (j * 16 + q);
    ape[idx] = (_Float16)x[src];
}

__global__ void k_cvt(const float* __restrict__ w, _Float16* __restrict__ o, int n) {
    int idx = blockIdx.x * 256 + threadIdx.x;
    if (idx < n) o[idx] = (_Float16)w[idx];
}

// w is [K,N] row-major f32 -> wt [N,K] row-major f16
__global__ void k_cvt_t(const float* __restrict__ w, _Float16* __restrict__ wt, int K, int N) {
    int idx = blockIdx.x * 256 + threadIdx.x;
    if (idx >= K * N) return;
    int nn = idx / K, kk = idx - nn * K;
    wt[idx] = (_Float16)w[(size_t)kk * N + nn];
}

// ---------------- async-LDS double-buffered f16 WMMA GEMM ----------------
// C[M,N] = A[M,K] * Bt[N,K]^T (+bias)(+relu)(+resid).
// Block tile 64x128, 4 waves (2x2), wave tile 32x64. M%64==0, N%128==0, K%32==0.
#define LDS_ROWB 80                   // padded row stride (64B data + 16B pad)
#define A_BYTES  (64 * LDS_ROWB)     // 5120
#define B_BYTES  (128 * LDS_ROWB)    // 10240
#define BUF_BYTES (A_BYTES + B_BYTES)

__global__ __launch_bounds__(128) void k_gemm(
    const _Float16* __restrict__ A, const _Float16* __restrict__ Bt,
    const float* __restrict__ bias, const float* __restrict__ resid,
    float* __restrict__ Cf, _Float16* __restrict__ Ch,
    int M, int N, int K, int relu)
{
    __shared__ __align__(16) char smem[2 * BUF_BYTES];
    const int tid = threadIdx.x;
    const int lane = tid & 31, wave = tid >> 5;
    const int half = lane >> 4, nl = lane & 15;
    const int mblk = blockIdx.y * 64, nblk = blockIdx.x * 128;
    const int m0w = (wave >> 1) * 32, n0w = (wave & 1) * 64;

    // issue one 32-K chunk of A(64 rows) + B(128 rows) into LDS buffer `buf`
    auto issue = [&](int kc, int buf) {
        const unsigned ldsbase = (unsigned)(size_t)(smem + buf * BUF_BYTES);
#pragma unroll
        for (int s = 0; s < 6; ++s) {
            const int gid = s * 128 + tid;   // 0..767 : 256 A segs then 512 B segs
            unsigned lds;
            const _Float16* gp;
            if (gid < 256) {
                const int row = gid >> 2, so = gid & 3;
                gp = A + (size_t)(mblk + row) * K + kc + so * 8;
                lds = ldsbase + row * LDS_ROWB + so * 16;
            } else {
                const int bg = gid - 256;
                const int row = bg >> 2, so = bg & 3;
                gp = Bt + (size_t)(nblk + row) * K + kc + so * 8;
                lds = ldsbase + A_BYTES + row * LDS_ROWB + so * 16;
            }
            unsigned long long ga = (unsigned long long)(size_t)gp;
            asm volatile("global_load_async_to_lds_b128 %0, %1, off"
                         :: "v"(lds), "v"(ga) : "memory");
        }
    };

    FragF c[2][4];
#pragma unroll
    for (int mt = 0; mt < 2; ++mt)
#pragma unroll
        for (int nt = 0; nt < 4; ++nt) c[mt][nt].v = (v8f){};

    const int nch = K >> 5;
    issue(0, 0);
    for (int i = 0; i < nch; ++i) {
        const int buf = i & 1;
        if (i + 1 < nch) {
            issue((i + 1) << 5, buf ^ 1);
            asm volatile("s_wait_asynccnt 0x6" ::: "memory"); // chunk i done
        } else {
            asm volatile("s_wait_asynccnt 0x0" ::: "memory");
        }
        __syncthreads();
        const char* base = smem + buf * BUF_BYTES;

        FragH a[2], b[4];
#pragma unroll
        for (int mt = 0; mt < 2; ++mt) {
            const uint4* p = (const uint4*)(base + (m0w + mt * 16 + nl) * LDS_ROWB);
            uint4 lo = p[2 * half], hi = p[2 * half + 1];
            a[mt].u[0] = lo.x; a[mt].u[1] = lo.y; a[mt].u[2] = lo.z; a[mt].u[3] = lo.w;
            a[mt].u[4] = hi.x; a[mt].u[5] = hi.y; a[mt].u[6] = hi.z; a[mt].u[7] = hi.w;
        }
#pragma unroll
        for (int nt = 0; nt < 4; ++nt) {
            const uint4* p = (const uint4*)(base + A_BYTES + (n0w + nt * 16 + nl) * LDS_ROWB + 32 * half);
            uint4 lo = p[0], hi = p[1];
            b[nt].u[0] = lo.x; b[nt].u[1] = lo.y; b[nt].u[2] = lo.z; b[nt].u[3] = lo.w;
            b[nt].u[4] = hi.x; b[nt].u[5] = hi.y; b[nt].u[6] = hi.z; b[nt].u[7] = hi.w;
        }
#pragma unroll
        for (int mt = 0; mt < 2; ++mt)
#pragma unroll
            for (int nt = 0; nt < 4; ++nt)
                c[mt][nt].v = wmma_f16(a[mt].v, b[nt].v, c[mt][nt].v);
        __syncthreads(); // everyone done reading buf before it is refilled
    }

#pragma unroll
    for (int mt = 0; mt < 2; ++mt)
#pragma unroll
        for (int nt = 0; nt < 4; ++nt) {
            const int col = nblk + n0w + nt * 16 + nl;
            const float bs = bias ? bias[col] : 0.0f;
#pragma unroll
            for (int r = 0; r < 8; ++r) {
                const int row = mblk + m0w + mt * 16 + r + 8 * half;
                float val = c[mt][nt].f[r] + bs;
                if (relu) val = fmaxf(val, 0.0f);
                const size_t idx = (size_t)row * N + col;
                if (resid) val += resid[idx];
                if (Cf) Cf[idx] = val;
                if (Ch) Ch[idx] = (_Float16)val;
            }
        }
}

// ---------------- LayerNorm: f32 rows -> f16 rows ----------------
__global__ __launch_bounds__(256) void k_ln(const float* __restrict__ x,
                                            const float* __restrict__ g,
                                            const float* __restrict__ b,
                                            _Float16* __restrict__ o)
{
    __shared__ float ss[256], s2[256];
    const int row = blockIdx.x, tid = threadIdx.x;
    const float* xr = x + (size_t)row * Ee;
    float v0 = xr[tid], v1 = xr[tid + 256], v2 = xr[tid + 512];
    ss[tid] = v0 + v1 + v2;
    s2[tid] = v0 * v0 + v1 * v1 + v2 * v2;
    __syncthreads();
    for (int s = 128; s > 0; s >>= 1) {
        if (tid < s) { ss[tid] += ss[tid + s]; s2[tid] += s2[tid + s]; }
        __syncthreads();
    }
    const float mu = ss[0] * (1.0f / Ee);
    const float var = s2[0] * (1.0f / Ee) - mu * mu;
    const float rs = rsqrtf(var + 1e-5f);
    _Float16* orow = o + (size_t)row * Ee;
#pragma unroll
    for (int t = 0; t < 3; ++t) {
        int e = tid + t * 256;
        orow[e] = (_Float16)((xr[e] - mu) * rs * g[e] + b[e]);
    }
}

// ---------------- flash attention (causal), 1 wave per (b,h,qtile) ----------------
// qkv: [Rr, 2304] f16, q|k|v at col offsets 0/768/1536. O: [Rr, 768] f16.
__global__ __launch_bounds__(32) void k_attn(const _Float16* __restrict__ qkv,
                                             _Float16* __restrict__ O)
{
    const int lane = threadIdx.x, half = lane >> 4, nl = lane & 15;
    const int i0 = blockIdx.x * 16, head = blockIdx.y, b = blockIdx.z;
    const size_t base = (size_t)b * Tt * QKVN + head * HDd;
    const _Float16* Qp = qkv + base;
    const _Float16* Kp = qkv + base + Ee;
    const _Float16* Vp = qkv + base + 2 * Ee;
    const float NEG_INF = -__builtin_inff();

    __shared__ __align__(16) _Float16 Vl[32 * 64];
    __shared__ __align__(16) _Float16 Pl[16 * 32];

    // Q fragments: 16 rows x 64 (two 16x32 A-frags)
    FragH qa[2];
    const unsigned* qrow = (const unsigned*)(Qp + (size_t)(i0 + nl) * QKVN);
#pragma unroll
    for (int d = 0; d < 2; ++d)
#pragma unroll
        for (int v = 0; v < 8; ++v) qa[d].u[v] = qrow[d * 16 + a_dw(v, half)];

    FragF o[4];
    float mprev[8], lsum[8];
#pragma unroll
    for (int t = 0; t < 4; ++t) o[t].v = (v8f){};
#pragma unroll
    for (int r = 0; r < 8; ++r) { mprev[r] = NEG_INF; lsum[r] = 0.0f; }

    const unsigned vlds = (unsigned)(size_t)&Vl[lane * 64];
    const int imax = i0 + 15;
    for (int j0 = 0; j0 <= imax; j0 += 32) {
        // async-prefetch V chunk (row clamp: masked cols have P==0 exactly)
        {
            int row = j0 + lane;
            if (row > Tt - 1) row = Tt - 1;
            unsigned long long ga = (unsigned long long)(size_t)(Vp + (size_t)row * QKVN);
#pragma unroll
            for (int s = 0; s < 8; ++s) {
                asm volatile("global_load_async_to_lds_b128 %0, %1, off"
                             :: "v"(vlds + s * 16), "v"(ga + (unsigned long long)(s * 16))
                             : "memory");
            }
        }
        // S = Q * K^T  (16 x 32 -> two C frags), overlapped with the V async copy
        FragF s[2];
#pragma unroll
        for (int nt = 0; nt < 2; ++nt) {
            const int krow = j0 + nt * 16 + nl;
            const bool valid = krow < Tt;
            const unsigned* kr = (const unsigned*)(Kp + (size_t)(valid ? krow : 0) * QKVN);
            v8f acc = (v8f){};
#pragma unroll
            for (int d = 0; d < 2; ++d) {
                FragH kb;
#pragma unroll
                for (int v = 0; v < 8; ++v) kb.u[v] = valid ? kr[d * 16 + b_dw(v, half)] : 0u;
                acc = wmma_f16(qa[d].v, kb.v, acc);
            }
            s[nt].v = acc;
        }
        // masked online softmax (row r+8*half lives in VGPR r, col = lane&15)
#pragma unroll
        for (int r = 0; r < 8; ++r) {
            const int row = i0 + r + 8 * half;
            const int c0 = j0 + nl, c1 = c0 + 16;
            float a0 = (c0 <= row) ? s[0].f[r] * ATTN_SCALE : NEG_INF;
            float a1 = (c1 <= row) ? s[1].f[r] * ATTN_SCALE : NEG_INF;
            float t = fmaxf(a0, a1);
#pragma unroll
            for (int st = 1; st < 16; st <<= 1) t = fmaxf(t, __shfl_xor(t, st, 32));
            const float mn = fmaxf(mprev[r], t);
            const float al = __expf(mprev[r] - mn);
            const float e0 = __expf(a0 - mn), e1 = __expf(a1 - mn);
            float sum = e0 + e1;
#pragma unroll
            for (int st = 1; st < 16; st <<= 1) sum += __shfl_xor(sum, st, 32);
            lsum[r] = lsum[r] * al + sum;
            mprev[r] = mn;
#pragma unroll
            for (int nt = 0; nt < 4; ++nt) o[nt].f[r] *= al;
            Pl[(r + 8 * half) * 32 + nl] = (_Float16)e0;
            Pl[(r + 8 * half) * 32 + 16 + nl] = (_Float16)e1;
        }
        asm volatile("s_wait_asynccnt 0x0" ::: "memory"); // V chunk resident
        __syncthreads();
        // P (C-layout in LDS) -> A-fragment
        FragH pa;
        const unsigned* pr = (const unsigned*)&Pl[nl * 32];
#pragma unroll
        for (int v = 0; v < 8; ++v) pa.u[v] = pr[a_dw(v, half)];
        // O += P * V
#pragma unroll
        for (int nt = 0; nt < 4; ++nt) {
            FragH vb;
#pragma unroll
            for (int v = 0; v < 8; ++v) {
                const int k = 2 * v + 16 * half;
                vb.h[2 * v]     = Vl[k * 64 + nt * 16 + nl];
                vb.h[2 * v + 1] = Vl[(k + 1) * 64 + nt * 16 + nl];
            }
            o[nt].v = wmma_f16(pa.v, vb.v, o[nt].v);
        }
        __syncthreads();
    }
#pragma unroll
    for (int nt = 0; nt < 4; ++nt)
#pragma unroll
        for (int r = 0; r < 8; ++r) {
            const int row = i0 + r + 8 * half;
            O[(size_t)b * Tt * Ee + (size_t)row * Ee + head * HDd + nt * 16 + nl] =
                (_Float16)(o[nt].f[r] / lsum[r]);
        }
}

// ---------------- (B,T,E) f32 -> (B,E,36,36) f32 ----------------
__global__ void k_out(const float* __restrict__ tok, float* __restrict__ out, int n) {
    int idx = blockIdx.x * 256 + threadIdx.x;
    if (idx >= n) return;
    int t = idx % Tt;
    int e = (idx / Tt) % Ee;
    int b = idx / (Tt * Ee);
    out[idx] = tok[((size_t)b * Tt + t) * Ee + e];
}

// ---------------- host orchestration ----------------
extern "C" void kernel_launch(void* const* d_in, const int* in_sizes, int n_in,
                              void* d_out, int out_size, void* d_ws, size_t ws_size,
                              hipStream_t stream) {
    (void)in_sizes; (void)n_in; (void)out_size; (void)ws_size;
    const float* x      = (const float*)d_in[0];
    const float* conv_w = (const float*)d_in[1];
    const float* conv_b = (const float*)d_in[2];
    const float* wq     = (const float*)d_in[3];
    const float* wk     = (const float*)d_in[4];
    const float* wv     = (const float*)d_in[5];
    const float* wo     = (const float*)d_in[6];
    const float* bo     = (const float*)d_in[7];
    const float* w1     = (const float*)d_in[8];
    const float* b1     = (const float*)d_in[9];
    const float* w2     = (const float*)d_in[10];
    const float* b2     = (const float*)d_in[11];
    const float* ln1_g  = (const float*)d_in[12];
    const float* ln1_b  = (const float*)d_in[13];
    const float* ln2_g  = (const float*)d_in[14];
    const float* ln2_b  = (const float*)d_in[15];
    float* out = (float*)d_out;

    char* ws = (char*)d_ws;
    size_t off = 0;
    auto alloc = [&](size_t bytes) -> void* {
        void* p = ws + off;
        off += (bytes + 255) & ~(size_t)255;
        return p;
    };
    float*    tok   = (float*)    alloc((size_t)Rr * Ee * 4);
    _Float16* ape   = (_Float16*) alloc((size_t)Rr * Ee * 2);   // A_pe, later attn out
    _Float16* hbuf  = (_Float16*) alloc((size_t)Rr * Ee * 2);   // LN1 & LN2 outputs
    _Float16* qkvb  = (_Float16*) alloc((size_t)Rr * QKVN * 2); // fused q|k|v
    _Float16* h3    = (_Float16*) alloc((size_t)Rr * Ff * 2);
    _Float16* wpe   = (_Float16*) alloc((size_t)Ee * Ee * 2);
    _Float16* wqkvT = (_Float16*) alloc((size_t)QKVN * Ee * 2); // [2304,768]
    _Float16* woT   = (_Float16*) alloc((size_t)Ee * Ee * 2);
    _Float16* w1T   = (_Float16*) alloc((size_t)Ee * Ff * 2);
    _Float16* w2T   = (_Float16*) alloc((size_t)Ff * Ee * 2);

    const int nAE = Rr * Ee;
    // 1. precision/layout prep
    k_patchify<<<(nAE + 255) / 256, 256, 0, stream>>>(x, ape, nAE);
    k_cvt<<<(Ee * Ee + 255) / 256, 256, 0, stream>>>(conv_w, wpe, Ee * Ee);
    k_cvt_t<<<(Ee * Ee + 255) / 256, 256, 0, stream>>>(wq, wqkvT, Ee, Ee);
    k_cvt_t<<<(Ee * Ee + 255) / 256, 256, 0, stream>>>(wk, wqkvT + (size_t)Ee * Ee, Ee, Ee);
    k_cvt_t<<<(Ee * Ee + 255) / 256, 256, 0, stream>>>(wv, wqkvT + (size_t)2 * Ee * Ee, Ee, Ee);
    k_cvt_t<<<(Ee * Ee + 255) / 256, 256, 0, stream>>>(wo, woT, Ee, Ee);
    k_cvt_t<<<(Ee * Ff + 255) / 256, 256, 0, stream>>>(w1, w1T, Ee, Ff);
    k_cvt_t<<<(Ff * Ee + 255) / 256, 256, 0, stream>>>(w2, w2T, Ff, Ee);

    // 2. patch embed GEMM -> tok (f32)
    k_gemm<<<dim3(Ee / 128, Rr / 64), 128, 0, stream>>>(
        ape, wpe, conv_b, nullptr, tok, nullptr, Rr, Ee, Ee, 0);
    // 3. LN1 -> hbuf (f16)
    k_ln<<<Rr, 256, 0, stream>>>(tok, ln1_g, ln1_b, hbuf);
    // 4. fused QKV projection (f16 out, [Rr,2304])
    k_gemm<<<dim3(QKVN / 128, Rr / 64), 128, 0, stream>>>(
        hbuf, wqkvT, nullptr, nullptr, nullptr, qkvb, Rr, QKVN, Ee, 0);
    // 5. causal flash attention -> ape (f16)
    k_attn<<<dim3(Tt / 16, Hh, Bc), 32, 0, stream>>>(qkvb, ape);
    // 6. output projection + residual -> tok
    k_gemm<<<dim3(Ee / 128, Rr / 64), 128, 0, stream>>>(
        ape, woT, bo, tok, tok, nullptr, Rr, Ee, Ee, 0);
    // 7. LN2 -> hbuf
    k_ln<<<Rr, 256, 0, stream>>>(tok, ln2_g, ln2_b, hbuf);
    // 8. MLP up + ReLU (f16 out)
    k_gemm<<<dim3(Ff / 128, Rr / 64), 128, 0, stream>>>(
        hbuf, w1T, b1, nullptr, nullptr, h3, Rr, Ff, Ee, 1);
    // 9. MLP down + residual -> tok
    k_gemm<<<dim3(Ee / 128, Rr / 64), 128, 0, stream>>>(
        h3, w2T, b2, tok, tok, nullptr, Rr, Ee, Ff, 0);
    // 10. (B,T,E) -> (B,E,36,36)
    k_out<<<(nAE + 255) / 256, 256, 0, stream>>>(tok, out, nAE);
}
